// Distribution_21921513079032
// MI455X (gfx1250) — compile-verified, matrix operand order checked by reading, not run on hardware
//
#include <hip/hip_runtime.h>

// Problem constants (from reference): B=2048, S=64, H=1024
#define HH 1024
#define BB 2048
#define SS 64
#define ROWS (BB * SS)     // 131072
#define KC 32              // K tile (WMMA K=32 for bf16)
#define MT 32              // rows per block
#define THREADS 512        // 16 waves (wave32)
#define LN_EPS 1e-5f

typedef __bf16 bf16;
typedef __attribute__((ext_vector_type(16))) bf16  v16bf;
typedef __attribute__((ext_vector_type(8)))  bf16  v8bf;
typedef __attribute__((ext_vector_type(8)))  float v8f;
typedef __attribute__((ext_vector_type(4)))  int   v4i;

union Frag { v16bf v; v8bf h[2]; };

// ---- gfx1250 async global->LDS copy path (ASYNCcnt-tracked), with fallback ----
#if defined(__has_builtin)
#if __has_builtin(__builtin_amdgcn_global_load_async_to_lds_b128)
#define ASYNC_B 1
#endif
#endif
#ifndef ASYNC_B
#define ASYNC_B 0
#endif

#if ASYNC_B
__device__ __forceinline__ void async_copy16(const bf16* g, bf16* l) {
  // Signature (from hipcc diagnostic): (v4i AS1*, v4i AS3*, imm int, imm int)
  __builtin_amdgcn_global_load_async_to_lds_b128(
      (__attribute__((address_space(1))) v4i*)g,
      (__attribute__((address_space(3))) v4i*)l, 0, 0);
}
__device__ __forceinline__ void wait_async0() {
#if __has_builtin(__builtin_amdgcn_s_wait_asynccnt)
  __builtin_amdgcn_s_wait_asynccnt(0);
#else
  asm volatile("s_wait_asynccnt 0x0" ::: "memory");
#endif
}
#endif

// Convert f32 HxH row-major -> bf16 HxH column-major (transposed), so that
// per-column K-runs are contiguous for 16B-wide B-tile staging + frag loads.
__global__ __launch_bounds__(256) void cvt_transpose_kernel(
    const float* __restrict__ src, bf16* __restrict__ dst) {
  int idx = blockIdx.x * 256 + threadIdx.x;   // 0 .. H*H-1
  int k = idx >> 10;
  int n = idx & (HH - 1);
  dst[(size_t)n * HH + k] = (bf16)src[idx];
}

// C[MT x H] = A[MT x H] @ W[H x H] + bias, then per-row LayerNorm over H.
// FUSED=false: out = LN * gamma + beta_ln written as (row,H) f32 (the query).
// FUSED=true : h = relu(query[b] + LN*gamma+beta_ln); out[row] = (h.Wb + bb)/32.
template <bool FUSED>
__global__ __launch_bounds__(THREADS) void gemm_ln_kernel(
    const float* __restrict__ A,       // rows x H, row-major
    const bf16*  __restrict__ Wt,      // H x H, column-major bf16 (pre-transposed)
    const float* __restrict__ bias,    // H
    const float* __restrict__ gamma,   // H
    const float* __restrict__ beta_ln, // H
    const float* __restrict__ query,   // B x H (FUSED only)
    const float* __restrict__ Wb,      // H (FUSED only)
    const float* __restrict__ bbp,     // scalar (FUSED only)
    float* __restrict__ out) {
  extern __shared__ char smem[];
  bf16* Bs0 = (bf16*)smem;                        // [n][k]: H x KC, buffer 0
  bf16* Bs1 = Bs0 + (size_t)KC * HH;              // buffer 1
  bf16* As0 = Bs1 + (size_t)KC * HH;              // [m][k]: MT x KC, buffer 0
  bf16* As1 = As0 + MT * KC;                      // buffer 1
  float* red_sum  = (float*)(As1 + MT * KC);
  float* red_sq   = red_sum + MT;
  float* red_beta = red_sq + MT;

  const int tid   = threadIdx.x;
  const int wave  = tid >> 5;
  const int lane  = tid & 31;
  const int col16 = lane & 15;
  const int g8    = (lane >> 4) * 8;   // K-half select per ISA 16-bit layout
  const int mg    = wave >> 3;         // 0..1  : which 16-row half
  const int ng    = wave & 7;          // 0..7  : 128-col slice
  const int nb    = ng * 128;
  const int row0  = blockIdx.x * MT;
  const int rowloc0 = mg * 16 + g8;    // first of the 8 C-rows this lane holds

  // B staging: 8 chunks of 8 bf16 per thread per step (all 16B transfers)
  const int nbase = tid >> 2;          // column base (stride 128 between chunks)
  const int kcoff = (tid & 3) * 8;     // K offset within KC
  // A staging: 2 f32 -> packed bf16x2 per thread per step
  const int am  = tid >> 4;
  const int akk = (tid & 15) * 2;

  if (tid < MT) { red_sum[tid] = 0.f; red_sq[tid] = 0.f; red_beta[tid] = 0.f; }

  v8f acc[8] = {};
#if !ASYNC_B
  uint4 bcarry[8];
#endif
  float2 acarry;

  // ---- prologue: start fetch of tile 0 into buffer 0 ----
  #pragma unroll
  for (int i = 0; i < 8; ++i) {
    const bf16* src = Wt + (size_t)(nbase + i * 128) * HH + kcoff;
#if ASYNC_B
    async_copy16(src, Bs0 + (nbase + i * 128) * KC + kcoff);
#else
    bcarry[i] = *(const uint4*)src;
#endif
  }
  acarry = *(const float2*)(A + (size_t)(row0 + am) * HH + akk);

  for (int it = 0; it < HH / KC; ++it) {
    bf16* Bcur = (it & 1) ? Bs1 : Bs0;
    bf16* Acur = (it & 1) ? As1 : As0;
    bf16* Bnxt = (it & 1) ? Bs0 : Bs1;

    // ---- commit tile `it` into LDS ----
#if ASYNC_B
    wait_async0();                 // our async b128 copies into Bcur are done
#else
    #pragma unroll
    for (int i = 0; i < 8; ++i)
      *(uint4*)(Bcur + (nbase + i * 128) * KC + kcoff) = bcarry[i];
#endif
    {
      union { bf16 h[2]; unsigned u; } pk;
      pk.h[0] = (bf16)acarry.x; pk.h[1] = (bf16)acarry.y;
      *(unsigned*)(Acur + am * KC + akk) = pk.u;
    }
    __syncthreads();               // tile `it` visible to all waves

    // ---- start fetch of tile it+1 (overlaps the WMMA compute below) ----
    const int k1 = (it + 1) * KC;
    if (k1 < HH) {
      #pragma unroll
      for (int i = 0; i < 8; ++i) {
        const bf16* src = Wt + (size_t)(nbase + i * 128) * HH + k1 + kcoff;
#if ASYNC_B
        async_copy16(src, Bnxt + (nbase + i * 128) * KC + kcoff);
#else
        bcarry[i] = *(const uint4*)src;
#endif
      }
      acarry = *(const float2*)(A + (size_t)(row0 + am) * HH + k1 + akk);
    }

    // ---- compute on tile `it` ----
    Frag fa;
    const bf16* apf = Acur + (mg * 16 + col16) * KC;
    fa.h[0] = *(const v8bf*)(apf + g8);
    fa.h[1] = *(const v8bf*)(apf + 16 + g8);
    #pragma unroll
    for (int t = 0; t < 8; ++t) {
      Frag fb;
      const bf16* bpf = Bcur + (nb + t * 16 + col16) * KC;
      fb.h[0] = *(const v8bf*)(bpf + g8);
      fb.h[1] = *(const v8bf*)(bpf + 16 + g8);
      acc[t] = __builtin_amdgcn_wmma_f32_16x16x32_bf16(
          false, fa.v, false, fb.v, (short)0, acc[t], false, false);
    }
    // No trailing barrier needed: each WMMA's s_wait_dscnt drains this wave's
    // frag loads before it reaches the next iteration's barrier, and writes in
    // iteration it+1 target the other buffer.
  }

  // ---- epilogue: bias add + per-row sum/sumsq partials ----
  float psum[8], psq[8];
  #pragma unroll
  for (int j = 0; j < 8; ++j) { psum[j] = 0.f; psq[j] = 0.f; }
  #pragma unroll
  for (int t = 0; t < 8; ++t) {
    int n = nb + t * 16 + col16;
    float bv = bias[n];
    #pragma unroll
    for (int j = 0; j < 8; ++j) {
      float c = acc[t][j] + bv;
      acc[t][j] = c;
      psum[j] += c;
      psq[j]  += c * c;
    }
  }
  #pragma unroll
  for (int j = 0; j < 8; ++j) {
    atomicAdd(&red_sum[rowloc0 + j], psum[j]);
    atomicAdd(&red_sq [rowloc0 + j], psq[j]);
  }
  __syncthreads();

  float mean[8], rstd[8];
  #pragma unroll
  for (int j = 0; j < 8; ++j) {
    float m = red_sum[rowloc0 + j] * (1.0f / HH);
    float v = red_sq [rowloc0 + j] * (1.0f / HH) - m * m;
    mean[j] = m;
    rstd[j] = rsqrtf(v + LN_EPS);
  }

  if (FUSED) {
    const int b = row0 >> 6;                      // row0 / S (block stays in one b)
    const float* qrow = query + (size_t)b * HH;
    float bacc[8];
    #pragma unroll
    for (int j = 0; j < 8; ++j) bacc[j] = 0.f;
    #pragma unroll
    for (int t = 0; t < 8; ++t) {
      int n = nb + t * 16 + col16;
      float gv = gamma[n], bv = beta_ln[n], wv = Wb[n], qv = qrow[n];
      #pragma unroll
      for (int j = 0; j < 8; ++j) {
        float ln = (acc[t][j] - mean[j]) * rstd[j] * gv + bv;
        float h  = fmaxf(qv + ln, 0.f);
        bacc[j] += h * wv;
      }
    }
    #pragma unroll
    for (int j = 0; j < 8; ++j) atomicAdd(&red_beta[rowloc0 + j], bacc[j]);
    __syncthreads();
    if (tid < MT)
      out[row0 + tid] = (red_beta[tid] + bbp[0]) * 0.03125f;  // /sqrt(1024)
  } else {
    #pragma unroll
    for (int t = 0; t < 8; ++t) {
      int n = nb + t * 16 + col16;
      float gv = gamma[n], bv = beta_ln[n];
      #pragma unroll
      for (int j = 0; j < 8; ++j) {
        float ln = (acc[t][j] - mean[j]) * rstd[j] * gv + bv;
        out[(size_t)(row0 + rowloc0 + j) * HH + n] = ln;
      }
    }
  }
}

// Masked cumulative softmax over S=64 per batch row -> (cp, rcp, p)
__global__ __launch_bounds__(SS) void softmax_kernel(
    const float* __restrict__ beta, const float* __restrict__ prev_p,
    float* __restrict__ cp_out, float* __restrict__ rcp_out,
    float* __restrict__ p_out) {
  __shared__ float buf[SS];
  __shared__ float scan[SS];
  __shared__ float sval;
  const int b = blockIdx.x;
  const int t = threadIdx.x;

  float bv = beta[b * SS + t];
  buf[t] = bv;
  __syncthreads();
  if (t == 0) {
    float m = buf[0];
    for (int i = 1; i < SS; ++i) m = fmaxf(m, buf[i]);
    sval = m;
  }
  __syncthreads();
  float x = expf(bv - sval);

  buf[t] = prev_p[b * SS + t];
  __syncthreads();
  if (t == 0) {
    float c = 0.f;
    for (int i = 0; i < SS; ++i) { c += buf[i]; scan[i] = c; }
  }
  __syncthreads();
  float m1 = (t < SS - 1) ? scan[t + 1] : 1.0f;
  float mask = (t < SS - 1) ? ((m1 < 1e-5f) ? 0.f : m1) : 1.0f;
  float xm = x * mask;
  buf[t] = fabsf(xm);
  __syncthreads();
  if (t == 0) {
    float s = 0.f;
    for (int i = 0; i < SS; ++i) s += buf[i];
    sval = fmaxf(s, 1e-12f);
  }
  __syncthreads();
  float p = xm / sval;
  buf[t] = p;
  __syncthreads();
  if (t == 0) {
    float c = 0.f;
    for (int i = 0; i < SS; ++i) { c += buf[i]; scan[i] = c; }
  }
  __syncthreads();
  float cp = scan[t];
  float total = scan[SS - 1];
  cp_out [b * SS + t] = cp;
  rcp_out[b * SS + t] = total - cp + p;   // reverse cumsum identity
  p_out  [b * SS + t] = p;
}

extern "C" void kernel_launch(void* const* d_in, const int* in_sizes, int n_in,
                              void* d_out, int out_size, void* d_ws, size_t ws_size,
                              hipStream_t stream) {
  const float* in_val = (const float*)d_in[0];
  const float* prevM  = (const float*)d_in[1];
  const float* prev_p = (const float*)d_in[2];
  const float* Wq     = (const float*)d_in[3];
  const float* bq     = (const float*)d_in[4];
  const float* gq     = (const float*)d_in[5];
  const float* betaq  = (const float*)d_in[6];
  const float* Wk     = (const float*)d_in[7];
  const float* bk     = (const float*)d_in[8];
  const float* gk     = (const float*)d_in[9];
  const float* betak  = (const float*)d_in[10];
  const float* Wb     = (const float*)d_in[11];
  const float* bb     = (const float*)d_in[12];

  char* ws = (char*)d_ws;
  bf16*  WqT      = (bf16*)ws;                                // 2 MB
  bf16*  WkT      = (bf16*)(ws + (size_t)2 * 1024 * 1024);    // 2 MB
  float* queryBuf = (float*)(ws + (size_t)4 * 1024 * 1024);   // 8 MB
  float* betaBuf  = (float*)(ws + (size_t)12 * 1024 * 1024);  // 512 KB

  cvt_transpose_kernel<<<(HH * HH) / 256, 256, 0, stream>>>(Wq, WqT);
  cvt_transpose_kernel<<<(HH * HH) / 256, 256, 0, stream>>>(Wk, WkT);

  size_t shmem = (size_t)2 * KC * HH * sizeof(bf16)   // B tiles (2 x 64 KB)
               + (size_t)2 * MT * KC * sizeof(bf16)   // A tiles (2 x 2 KB)
               + (size_t)3 * MT * sizeof(float);      // reductions

  // query = LN(in_val @ Wq + bq)*gq + betaq      (2048 rows -> 64 blocks)
  gemm_ln_kernel<false><<<BB / MT, THREADS, shmem, stream>>>(
      in_val, WqT, bq, gq, betaq, nullptr, nullptr, nullptr, queryBuf);

  // beta = (relu(query + LN(prevM @ Wk + bk)*gk + betak) . Wb + bb) / 32
  gemm_ln_kernel<true><<<ROWS / MT, THREADS, shmem, stream>>>(
      prevM, WkT, bk, gk, betak, queryBuf, Wb, bb, betaBuf);

  float* cp  = (float*)d_out;
  float* rcp = cp + (size_t)BB * SS;
  float* pO  = rcp + (size_t)BB * SS;
  softmax_kernel<<<BB, SS, 0, stream>>>(betaBuf, prev_p, cp, rcp, pO);
}
